// BDART_3307124818408
// MI455X (gfx1250) — matrix operand order repaired and checked by reading, not compiled, not run on hardware
//
#include <hip/hip_runtime.h>
#include <hip/hip_bf16.h>

typedef __bf16 v16bf __attribute__((ext_vector_type(16)));
typedef __bf16 v8bf  __attribute__((ext_vector_type(8)));
typedef float  v8f   __attribute__((ext_vector_type(8)));
typedef unsigned short ushort_t;

#define S_DIM 128
#define H_DIM 4096
#define A_DIM 4
#define B_DIM 4096
#define NOUT  2048   // S*A*A

// ---------- f32 -> bf16 (round-to-nearest-even) ----------
__device__ __forceinline__ ushort_t f2bf(float f) {
    unsigned u = __float_as_uint(f);
    unsigned r = u + 0x7FFFu + ((u >> 16) & 1u);
    return (ushort_t)(r >> 16);
}

// ---------- conversion + masking kernels ----------
__global__ void cvt_x_k(const float* __restrict__ X, ushort_t* __restrict__ out, int n) {
    int i = blockIdx.x * blockDim.x + threadIdx.x;
    if (i < n) out[i] = f2bf(X[i]);
}
// W0: (H, S); mask: (n % 127) >= k
__global__ void cvt_w0_k(const float* __restrict__ W, ushort_t* __restrict__ out) {
    int i = blockIdx.x * blockDim.x + threadIdx.x;        // H*S = 524288
    int nrow = i >> 7, k = i & 127;
    float w = ((nrow % 127) >= k) ? W[i] : 0.0f;
    out[i] = f2bf(w);
}
// W1/2/3: (H, H); mask: (n % 127) >= (k % 127)
__global__ void cvt_wh_k(const float* __restrict__ W, ushort_t* __restrict__ out) {
    int i = blockIdx.x * blockDim.x + threadIdx.x;        // H*H = 16777216
    int nrow = i >> 12, k = i & 4095;
    float w = ((nrow % 127) >= (k % 127)) ? W[i] : 0.0f;
    out[i] = f2bf(w);
}
// Wout: (S*A*A, H); row r -> s = r/16; mask: (s-1) >= (k % 127)  (signed)
__global__ void cvt_wout_k(const float* __restrict__ W, ushort_t* __restrict__ out) {
    int i = blockIdx.x * blockDim.x + threadIdx.x;        // 2048*4096 = 8388608
    int r = i >> 12, k = i & 4095;
    int s = r >> 4;
    float w = ((s - 1) >= (k % 127)) ? W[i] : 0.0f;
    out[i] = f2bf(w);
}

// ---------- tiled WMMA GEMM:  C[M,N] = act(A[M,K] @ B[N,K]^T + bias[N]) ----------
// Block tile 256x128, 8 waves (256 thr) arranged 4x2, wave tile 64x64, K-step 32.
// Per K-step per wave: 8 ds_load_b128 (A) + 8 ds_load_b128 (B) -> 16 v_wmma (1:1).
#define LDS_STR 40   // 32 halfs + 8 pad (80B = 20 dwords: conflict-free b128 reads)
#define BM 256
#define BN 128

template <bool RELU, bool F32OUT>
__global__ __launch_bounds__(256) void gemm_bf16(
    const ushort_t* __restrict__ Ag, const ushort_t* __restrict__ Bg,
    const float* __restrict__ bias, void* __restrict__ Cg,
    int N, int K)
{
    __shared__ __align__(16) ushort_t As[BM * LDS_STR];
    __shared__ __align__(16) ushort_t Bs[BN * LDS_STR];

    const int tid  = threadIdx.x;
    const int lane = tid & 31;
    const int wave = tid >> 5;
    const int m0w  = (wave >> 1) * 64;   // 4 wave-rows
    const int n0w  = (wave & 1) * 64;    // 2 wave-cols
    const size_t blockM = (size_t)blockIdx.y * BM;
    const size_t blockN = (size_t)blockIdx.x * BN;

    // A staging: thread t loads full 32-half row t (4 x b128)
    // B staging: thread t loads 16 halfs of row t/2, chunk (t&1)*16 (2 x b128)
    const int brow = tid >> 1;
    const int bcol = (tid & 1) * 16;

    v8f acc[4][4] = {};

    for (int k0 = 0; k0 < K; k0 += 32) {
        const ushort_t* ap = Ag + (blockM + tid) * (size_t)K + k0;
        const ushort_t* bp = Bg + (blockN + brow) * (size_t)K + k0 + bcol;
        uint4 av0 = *(const uint4*)(ap);
        uint4 av1 = *(const uint4*)(ap + 8);
        uint4 av2 = *(const uint4*)(ap + 16);
        uint4 av3 = *(const uint4*)(ap + 24);
        uint4 bv0 = *(const uint4*)(bp);
        uint4 bv1 = *(const uint4*)(bp + 8);
        if (k0 + 32 < K) {                     // prefetch next K-chunk
            __builtin_prefetch(ap + 32, 0, 1);
            __builtin_prefetch(bp + 32, 0, 1);
        }
        __syncthreads();
        *(uint4*)&As[tid * LDS_STR]      = av0;
        *(uint4*)&As[tid * LDS_STR + 8]  = av1;
        *(uint4*)&As[tid * LDS_STR + 16] = av2;
        *(uint4*)&As[tid * LDS_STR + 24] = av3;
        *(uint4*)&Bs[brow * LDS_STR + bcol]     = bv0;
        *(uint4*)&Bs[brow * LDS_STR + bcol + 8] = bv1;
        __syncthreads();

        // A fragments: lane L -> row L%16; halves 0-7 = K kb..kb+7, 8-15 = kb+16..kb+23, kb=(L/16)*8
        const int kbA = (lane >> 4) * 8;
        v16bf afrag[4];
        #pragma unroll
        for (int mi = 0; mi < 4; ++mi) {
            const int r = m0w + mi * 16 + (lane & 15);
            v8bf lo = *(const v8bf*)&As[r * LDS_STR + kbA];
            v8bf hi = *(const v8bf*)&As[r * LDS_STR + kbA + 16];
            afrag[mi] = __builtin_shufflevector(lo, hi, 0,1,2,3,4,5,6,7,8,9,10,11,12,13,14,15);
        }
        // B fragments: lane L -> col L%16; halves 0-7 = K kb..kb+7, 8-15 = kb+16..kb+23, kb=(L/16)*8
        const int kbB = (lane >> 4) * 8;
        v16bf bfrag[4];
        #pragma unroll
        for (int ni = 0; ni < 4; ++ni) {
            const int r = n0w + ni * 16 + (lane & 15);
            v8bf lo = *(const v8bf*)&Bs[r * LDS_STR + kbB];
            v8bf hi = *(const v8bf*)&Bs[r * LDS_STR + kbB + 16];
            bfrag[ni] = __builtin_shufflevector(lo, hi, 0,1,2,3,4,5,6,7,8,9,10,11,12,13,14,15);
        }
        // 16 back-to-back WMMAs
        #pragma unroll
        for (int ni = 0; ni < 4; ++ni) {
            #pragma unroll
            for (int mi = 0; mi < 4; ++mi) {
                acc[mi][ni] = __builtin_amdgcn_wmma_f32_16x16x32_bf16(
                    false, afrag[mi], false, bfrag[ni], (short)0, acc[mi][ni], false, false);
            }
        }
    }

    // Epilogue: C layout VGPR v, lane L -> (M=(L/16)*8+v, N=L%16)
    #pragma unroll
    for (int mi = 0; mi < 4; ++mi) {
        #pragma unroll
        for (int ni = 0; ni < 4; ++ni) {
            const size_t nn = blockN + n0w + ni * 16 + (lane & 15);
            const float bv = bias[nn];
            #pragma unroll
            for (int v = 0; v < 8; ++v) {
                const size_t mm = blockM + m0w + mi * 16 + ((lane >> 4) * 8) + v;
                float val = acc[mi][ni][v] + bv;
                if (RELU) val = fmaxf(val, 0.0f);
                if (F32OUT) ((float*)Cg)[mm * (size_t)N + nn] = val;
                else        ((ushort_t*)Cg)[mm * (size_t)N + nn] = f2bf(val);
            }
        }
    }
}

// ---------- log-space chain: one thread per batch ----------
__device__ __forceinline__ float logm_elem(float t, float xs) {
    // xs*t - softplus(t) + log(1/4), softplus stable
    const float LOGQ = -1.3862943611198906f;
    float sp = fmaxf(t, 0.0f) + log1pf(expf(-fabsf(t)));
    return __fmaf_rn(xs, t, -sp) + LOGQ;
}

__global__ __launch_bounds__(256) void chain_k(const float* __restrict__ theta,
                                               const float* __restrict__ x,
                                               float* __restrict__ out)
{
    int b = blockIdx.x * blockDim.x + threadIdx.x;
    if (b >= B_DIM) return;
    const float* th = theta + (size_t)b * NOUT;
    const float* xb = x + (size_t)b * S_DIM;

    float v[4];
    {   // first = logm[s=0] row 0
        float xs = xb[0];
        #pragma unroll
        for (int j = 0; j < 4; ++j) v[j] = logm_elem(th[j], xs);
    }
    for (int s = 1; s <= S_DIM - 2; ++s) {
        float xs = xb[s];
        const float4* t4 = (const float4*)(th + s * 16);
        float mm[16];
        #pragma unroll
        for (int k = 0; k < 4; ++k) {
            float4 q = t4[k];
            mm[k*4+0] = logm_elem(q.x, xs);
            mm[k*4+1] = logm_elem(q.y, xs);
            mm[k*4+2] = logm_elem(q.z, xs);
            mm[k*4+3] = logm_elem(q.w, xs);
        }
        float nv[4];
        #pragma unroll
        for (int j = 0; j < 4; ++j) {
            float t0 = v[0] + mm[0*4+j], t1 = v[1] + mm[1*4+j];
            float t2 = v[2] + mm[2*4+j], t3 = v[3] + mm[3*4+j];
            float mx = fmaxf(fmaxf(t0, t1), fmaxf(t2, t3));
            nv[j] = mx + logf(expf(t0-mx) + expf(t1-mx) + expf(t2-mx) + expf(t3-mx));
        }
        v[0]=nv[0]; v[1]=nv[1]; v[2]=nv[2]; v[3]=nv[3];
    }
    {   // last = logm[s=127] column 0
        float xs = xb[S_DIM - 1];
        float tk[4];
        #pragma unroll
        for (int k = 0; k < 4; ++k)
            tk[k] = v[k] + logm_elem(th[(S_DIM - 1) * 16 + k * 4], xs);
        float mx = fmaxf(fmaxf(tk[0], tk[1]), fmaxf(tk[2], tk[3]));
        out[b] = mx + logf(expf(tk[0]-mx) + expf(tk[1]-mx) + expf(tk[2]-mx) + expf(tk[3]-mx));
    }
}

// ---------- host launch ----------
extern "C" void kernel_launch(void* const* d_in, const int* in_sizes, int n_in,
                              void* d_out, int out_size, void* d_ws, size_t ws_size,
                              hipStream_t stream) {
    (void)in_sizes; (void)n_in; (void)out_size; (void)ws_size;
    const float* x    = (const float*)d_in[0];
    const float* W0   = (const float*)d_in[1];
    const float* b0   = (const float*)d_in[2];
    const float* W1   = (const float*)d_in[3];
    const float* b1   = (const float*)d_in[4];
    const float* W2   = (const float*)d_in[5];
    const float* b2   = (const float*)d_in[6];
    const float* W3   = (const float*)d_in[7];
    const float* b3   = (const float*)d_in[8];
    const float* Wout = (const float*)d_in[9];
    const float* bout = (const float*)d_in[10];

    char* ws = (char*)d_ws;
    size_t off = 0;
    auto carve = [&](size_t bytes) -> char* {
        char* p = ws + off;
        off += (bytes + 255) & ~(size_t)255;
        return p;
    };
    ushort_t* xbf  = (ushort_t*)carve((size_t)B_DIM * S_DIM * 2);
    ushort_t* w0b  = (ushort_t*)carve((size_t)H_DIM * S_DIM * 2);
    ushort_t* w1b  = (ushort_t*)carve((size_t)H_DIM * H_DIM * 2);
    ushort_t* w2b  = (ushort_t*)carve((size_t)H_DIM * H_DIM * 2);
    ushort_t* w3b  = (ushort_t*)carve((size_t)H_DIM * H_DIM * 2);
    ushort_t* wob  = (ushort_t*)carve((size_t)NOUT  * H_DIM * 2);
    ushort_t* hA   = (ushort_t*)carve((size_t)B_DIM * H_DIM * 2);
    ushort_t* hB   = (ushort_t*)carve((size_t)B_DIM * H_DIM * 2);
    float*    thet = (float*)   carve((size_t)B_DIM * NOUT  * 4);

    const int T = 256;
    cvt_x_k   <<<(B_DIM * S_DIM) / T, T, 0, stream>>>(x, xbf, B_DIM * S_DIM);
    cvt_w0_k  <<<(H_DIM * S_DIM) / T, T, 0, stream>>>(W0, w0b);
    cvt_wh_k  <<<(H_DIM * H_DIM) / T, T, 0, stream>>>(W1, w1b);
    cvt_wh_k  <<<(H_DIM * H_DIM) / T, T, 0, stream>>>(W2, w2b);
    cvt_wh_k  <<<(H_DIM * H_DIM) / T, T, 0, stream>>>(W3, w3b);
    cvt_wout_k<<<(NOUT  * H_DIM) / T, T, 0, stream>>>(Wout, wob);

    dim3 gHH(H_DIM / BN, B_DIM / BM);   // N/128, M/256
    gemm_bf16<true,  false><<<gHH, T, 0, stream>>>(xbf, w0b, b0, hA, H_DIM, S_DIM);
    gemm_bf16<true,  false><<<gHH, T, 0, stream>>>(hA,  w1b, b1, hB, H_DIM, H_DIM);
    gemm_bf16<true,  false><<<gHH, T, 0, stream>>>(hB,  w2b, b2, hA, H_DIM, H_DIM);
    gemm_bf16<true,  false><<<gHH, T, 0, stream>>>(hA,  w3b, b3, hB, H_DIM, H_DIM);
    dim3 gOut(NOUT / BN, B_DIM / BM);
    gemm_bf16<false, true ><<<gOut, T, 0, stream>>>(hB,  wob, bout, thet, NOUT, H_DIM);

    chain_k<<<B_DIM / T, T, 0, stream>>>(thet, x, (float*)d_out);
}
